// SelfAttention_12094627906139
// MI455X (gfx1250) — matrix operand order repaired and checked by reading, not compiled, not run on hardware
//
#include <hip/hip_runtime.h>

// ---------------------------------------------------------------------------
// Self-attention (QKV proj + softmax attention) for MI455X / gfx1250.
// All three matmuls (QKV GEMM, Q·K^T, P·V) run on v_wmma_f32_16x16x32_bf16.
// B=2, L=2048, D=1024, H=16, HD=64.
// Attention: 32 queries/wave, K/V staged in double-buffered LDS shared by the
// whole block (4x less L2 traffic than per-wave streaming).
// ---------------------------------------------------------------------------

typedef __bf16 bf16;
typedef __bf16 bf16x4  __attribute__((ext_vector_type(4)));
typedef __bf16 bf16x8  __attribute__((ext_vector_type(8)));
typedef __bf16 bf16x16 __attribute__((ext_vector_type(16)));
typedef float  v8f     __attribute__((ext_vector_type(8)));

#define WMMA_BF16(a, b, c) \
  __builtin_amdgcn_wmma_f32_16x16x32_bf16(false, (a), false, (b), (short)0, (c), false, false)

// A-fragment (16x32 bf16): lane = lr + 16*lh. element e -> K = 16*lh + e.
// `row` = this lane's row (K-major).
__device__ __forceinline__ bf16x16 load_afrag(const bf16* __restrict__ row, int koff, int lh) {
  bf16x8 lo = *(const bf16x8*)(row + koff + 16 * lh);
  bf16x8 hi = *(const bf16x8*)(row + koff + 16 * lh + 8);
  return __builtin_shufflevector(lo, hi, 0,1,2,3,4,5,6,7,8,9,10,11,12,13,14,15);
}

// B-fragment (32x16 bf16): lane = lc + 16*lh. element e -> K = 16*lh + e.
__device__ __forceinline__ bf16x16 load_bfrag(const bf16* __restrict__ row, int koff, int lh) {
  return *(const bf16x16*)(row + koff + 16 * lh);
}

__device__ __forceinline__ unsigned pack_bf16(float a, float b) {
  union { bf16 h; unsigned short s; } ua, ub;
  ua.h = (bf16)a; ub.h = (bf16)b;
  return (unsigned)ua.s | ((unsigned)ub.s << 16);
}

// ---------------------------------------------------------------------------
// 1) f32 -> bf16 conversion (4-wide)
// ---------------------------------------------------------------------------
__global__ __launch_bounds__(256) void cvt_f32_to_bf16(const float4* __restrict__ in,
                                                       bf16* __restrict__ out, int n4) {
  int i = blockIdx.x * 256 + threadIdx.x;
  if (i < n4) {
    float4 v = in[i];
    bf16x4 o = { (bf16)v.x, (bf16)v.y, (bf16)v.z, (bf16)v.w };
    *(bf16x4*)(out + 4 * (size_t)i) = o;
  }
}

// ---------------------------------------------------------------------------
// 2) QKV GEMM: C[m,n] = sum_k x[m,k] * W[n,k] + bias[n]
//    M=4096, N=3072, K=1024. Block = 8 waves (64x256 tile); wave tile = 32x64.
//    Writes Q,K as [B,H,L,64] bf16 and V transposed as [B,H,64,L] bf16.
// ---------------------------------------------------------------------------
__global__ __launch_bounds__(256) void qkv_gemm_wmma(const bf16* __restrict__ xb,
                                                     const bf16* __restrict__ wb,
                                                     const float* __restrict__ bias,
                                                     bf16* __restrict__ Qb,
                                                     bf16* __restrict__ Kb,
                                                     bf16* __restrict__ Vt) {
  const int lane = threadIdx.x & 31;
  const int wid  = threadIdx.x >> 5;       // 0..7
  const int wm = wid & 1, wn = wid >> 1;   // 2 waves in M, 4 in N
  const int lr = lane & 15, lh = lane >> 4;
  const int mbase = blockIdx.x * 64 + wm * 32;
  const int nbase = blockIdx.y * 256 + wn * 64;

  const bf16* arow0 = xb + (size_t)(mbase + lr) * 1024;
  const bf16* arow1 = xb + (size_t)(mbase + 16 + lr) * 1024;
  const bf16* brow[4];
#pragma unroll
  for (int nt = 0; nt < 4; ++nt) brow[nt] = wb + (size_t)(nbase + nt * 16 + lr) * 1024;

  v8f acc[2][4];
#pragma unroll
  for (int mt = 0; mt < 2; ++mt)
#pragma unroll
    for (int nt = 0; nt < 4; ++nt) acc[mt][nt] = (v8f){};

  for (int k0 = 0; k0 < 1024; k0 += 32) {
    bf16x16 a0 = load_afrag(arow0, k0, lh);
    bf16x16 a1 = load_afrag(arow1, k0, lh);
#pragma unroll
    for (int nt = 0; nt < 4; ++nt) {
      bf16x16 b = load_bfrag(brow[nt], k0, lh);
      acc[0][nt] = WMMA_BF16(a0, b, acc[0][nt]);
      acc[1][nt] = WMMA_BF16(a1, b, acc[1][nt]);
    }
  }

  // Epilogue: C layout -> lane column n = lr, row m = r + 8*lh (per 16x16 tile).
#pragma unroll
  for (int nt = 0; nt < 4; ++nt) {
    int n_g  = nbase + nt * 16 + lr;
    float bv = bias[n_g];
    int sec  = n_g >> 10;        // 0=Q, 1=K, 2=V
    int rem  = n_g & 1023;
    int head = rem >> 6, hd = rem & 63;
#pragma unroll
    for (int mt = 0; mt < 2; ++mt) {
#pragma unroll
      for (int r = 0; r < 8; ++r) {
        int m_g  = mbase + mt * 16 + r + 8 * lh;
        int bidx = m_g >> 11, l = m_g & 2047;
        bf16 v = (bf16)(acc[mt][nt][r] + bv);
        size_t bh = (size_t)(bidx * 16 + head);
        if (sec == 0)      Qb[(bh * 2048 + l) * 64 + hd] = v;
        else if (sec == 1) Kb[(bh * 2048 + l) * 64 + hd] = v;
        else               Vt[(bh * 64 + hd) * 2048 + l] = v;   // V stored transposed
      }
    }
  }
}

// ---------------------------------------------------------------------------
// 3) Flash attention. 4 waves/block, 32 queries/wave (2 groups of 16),
//    key chunks of 32 staged in double-buffered LDS shared by the block.
//    S^T = K · Q^T   (rows = keys, cols = queries -> softmax stats per-lane)
//    out^T = V^T · P^T (P^T built from S^T regs with one half-wave swap)
// ---------------------------------------------------------------------------
#define LK 72   // padded LDS row stride for K tile (elements): 36 dwords, conflict-free
#define LV 40   // padded LDS row stride for V tile (elements): 20 dwords, conflict-free

struct Stage { bf16x8 k0, k1, v0, v1; };

__device__ __forceinline__ Stage stage_load(const bf16* __restrict__ Kg,
                                            const bf16* __restrict__ Vg,
                                            int kc, int t) {
  Stage s;
  int kr = t >> 2, kcol = (t & 3) * 16;          // K tile: 32 rows x 64
  const bf16x8* gk = (const bf16x8*)(Kg + (size_t)(kc + kr) * 64 + kcol);
  s.k0 = gk[0]; s.k1 = gk[1];
  int vr = t >> 1, vcol = (t & 1) * 16;          // V tile: 64 rows x 32
  const bf16x8* gv = (const bf16x8*)(Vg + (size_t)vr * 2048 + kc + vcol);
  s.v0 = gv[0]; s.v1 = gv[1];
  return s;
}

__device__ __forceinline__ void stage_store(const Stage& s, bf16* skb, bf16* svb, int t) {
  int kr = t >> 2, kcol = (t & 3) * 16;
  bf16x8* dk = (bf16x8*)(skb + kr * LK + kcol);
  dk[0] = s.k0; dk[1] = s.k1;
  int vr = t >> 1, vcol = (t & 1) * 16;
  bf16x8* dv = (bf16x8*)(svb + vr * LV + vcol);
  dv[0] = s.v0; dv[1] = s.v1;
}

__global__ __launch_bounds__(128) void attn_wmma(const bf16* __restrict__ Qb,
                                                 const bf16* __restrict__ Kb,
                                                 const bf16* __restrict__ Vt,
                                                 float* __restrict__ out) {
  __shared__ __attribute__((aligned(16))) bf16 sK[2][32 * LK];
  __shared__ __attribute__((aligned(16))) bf16 sV[2][64 * LV];

  const int tid  = threadIdx.x;
  const int lane = tid & 31;
  const int wid  = tid >> 5;               // 0..3
  const int lr = lane & 15, lh = lane >> 4;
  const int bh   = blockIdx.x >> 4;        // (b*16 + head), 0..31
  const int qblk = blockIdx.x & 15;
  const int qbase = qblk * 128 + wid * 32; // wave covers queries [qbase, qbase+32)

  const size_t bhoff = (size_t)bh * 2048 * 64;
  const bf16* Q = Qb + bhoff;
  const bf16* K = Kb + bhoff;
  const bf16* V = Vt + bhoff;              // [64][2048]

  // Q^T B-fragments (column = query lr), loop-invariant; 2 query groups.
  const bf16* qrow0 = Q + (size_t)(qbase + lr) * 64;
  const bf16* qrow1 = Q + (size_t)(qbase + 16 + lr) * 64;
  bf16x16 qfa[2] = { load_bfrag(qrow0, 0, lh),  load_bfrag(qrow1, 0, lh)  };
  bf16x16 qfb[2] = { load_bfrag(qrow0, 32, lh), load_bfrag(qrow1, 32, lh) };

  v8f o[2][4];                             // out^T tiles per group: hd 0-15..48-63
#pragma unroll
  for (int g = 0; g < 2; ++g)
#pragma unroll
    for (int i = 0; i < 4; ++i) o[g][i] = (v8f){};
  float mrun[2] = { -1e30f, -1e30f };
  float lrun[2] = { 0.0f, 0.0f };
  const float sc = 0.18033688011112042f;   // log2(e) / sqrt(64)

  // Prologue: stage chunk 0.
  Stage st = stage_load(K, V, 0, tid);
  stage_store(st, sK[0], sV[0], tid);
  __syncthreads();

  int ib = 0;
  for (int kc = 0; kc < 2048; kc += 32, ib ^= 1) {
    const bool more = (kc + 32) < 2048;
    Stage nx = {};
    if (more) nx = stage_load(K, V, kc + 32, tid);   // prefetch next chunk

    const bf16* kb = sK[ib];
    const bf16* vb = sV[ib];

    // K A-fragments for both key tiles, both hd halves (shared by both groups).
    bf16x16 ka00 = load_afrag(kb + lr * LK,        0,  lh);
    bf16x16 ka01 = load_afrag(kb + lr * LK,        32, lh);
    bf16x16 ka10 = load_afrag(kb + (16 + lr) * LK, 0,  lh);
    bf16x16 ka11 = load_afrag(kb + (16 + lr) * LK, 32, lh);

    bf16x16 pf[2];
#pragma unroll
    for (int g = 0; g < 2; ++g) {
      v8f s0 = {}, s1 = {};
      s0 = WMMA_BF16(ka00, qfa[g], s0);
      s0 = WMMA_BF16(ka01, qfb[g], s0);
      s1 = WMMA_BF16(ka10, qfa[g], s1);
      s1 = WMMA_BF16(ka11, qfb[g], s1);
      s0 = s0 * sc; s1 = s1 * sc;          // log2-domain scaled scores

      float cmax = s0[0];
#pragma unroll
      for (int r = 1; r < 8; ++r) cmax = fmaxf(cmax, s0[r]);
#pragma unroll
      for (int r = 0; r < 8; ++r) cmax = fmaxf(cmax, s1[r]);
      cmax = fmaxf(cmax, __shfl_xor(cmax, 16, 32));
      float mnew  = fmaxf(mrun[g], cmax);
      float alpha = exp2f(mrun[g] - mnew);
      float p0[8], p1[8], csum = 0.0f;
#pragma unroll
      for (int r = 0; r < 8; ++r) { p0[r] = exp2f(s0[r] - mnew); csum += p0[r]; }
#pragma unroll
      for (int r = 0; r < 8; ++r) { p1[r] = exp2f(s1[r] - mnew); csum += p1[r]; }
      csum += __shfl_xor(csum, 16, 32);
      lrun[g] = lrun[g] * alpha + csum;
      mrun[g] = mnew;
#pragma unroll
      for (int i = 0; i < 4; ++i) o[g][i] = o[g][i] * alpha;

      // Build P^T B-fragment: own half keeps its K-block; other via shfl_xor(16).
      unsigned t0[4], t1[4], pt0[4], pt1[4];
#pragma unroll
      for (int j = 0; j < 4; ++j) {
        t0[j] = pack_bf16(p0[2 * j], p0[2 * j + 1]);
        t1[j] = pack_bf16(p1[2 * j], p1[2 * j + 1]);
      }
#pragma unroll
      for (int j = 0; j < 4; ++j) {
        pt0[j] = (unsigned)__shfl_xor((int)t0[j], 16, 32);
        pt1[j] = (unsigned)__shfl_xor((int)t1[j], 16, 32);
      }
      union { unsigned u[8]; bf16x16 v; } fu;
#pragma unroll
      for (int j = 0; j < 4; ++j) {
        fu.u[j]     = lh ? pt1[j] : t0[j];  // keys kc + 16*lh + 0..7
        fu.u[j + 4] = lh ? t1[j]  : pt0[j]; // keys kc + 16*lh + 8..15
      }
      pf[g] = fu.v;
    }

    // out^T += V^T x P^T : V fragments shared by both query groups.
#pragma unroll
    for (int i = 0; i < 4; ++i) {
      bf16x16 vf = load_afrag(vb + (i * 16 + lr) * LV, 0, lh);
      o[0][i] = WMMA_BF16(vf, pf[0], o[0][i]);
      o[1][i] = WMMA_BF16(vf, pf[1], o[1][i]);
    }

    __syncthreads();                        // everyone done reading buffer ib^1's successor slot
    if (more) stage_store(nx, sK[ib ^ 1], sV[ib ^ 1], tid);
    __syncthreads();                        // staged data visible before next compute
  }

  const int b = bh >> 4, hh = bh & 15;
#pragma unroll
  for (int g = 0; g < 2; ++g) {
    float inv = 1.0f / lrun[g];
    float* orow = out + (size_t)(b * 2048 + qbase + g * 16 + lr) * 1024 + hh * 64;
#pragma unroll
    for (int r = 0; r < 8; ++r) {
      orow[ 0 + r + 8 * lh] = o[g][0][r] * inv;
      orow[16 + r + 8 * lh] = o[g][1][r] * inv;
      orow[32 + r + 8 * lh] = o[g][2][r] * inv;
      orow[48 + r + 8 * lh] = o[g][3][r] * inv;
    }
  }
}

// ---------------------------------------------------------------------------
extern "C" void kernel_launch(void* const* d_in, const int* in_sizes, int n_in,
                              void* d_out, int out_size, void* d_ws, size_t ws_size,
                              hipStream_t stream) {
  (void)in_sizes; (void)n_in; (void)out_size; (void)ws_size;
  const float* x    = (const float*)d_in[0];
  const float* Wqkv = (const float*)d_in[1];
  const float* bqkv = (const float*)d_in[2];
  float* out = (float*)d_out;

  char* ws = (char*)d_ws;
  const size_t SZ_XB = (size_t)4096 * 1024 * 2;        // x bf16:  8 MiB
  const size_t SZ_WB = (size_t)3072 * 1024 * 2;        // W bf16:  6 MiB
  const size_t SZ_H  = (size_t)2 * 16 * 2048 * 64 * 2; // Q/K/Vt:  8 MiB each
  bf16* xb = (bf16*)ws;
  bf16* wb = (bf16*)(ws + SZ_XB);
  bf16* Qb = (bf16*)(ws + SZ_XB + SZ_WB);
  bf16* Kb = (bf16*)(ws + SZ_XB + SZ_WB + SZ_H);
  bf16* Vt = (bf16*)(ws + SZ_XB + SZ_WB + 2 * SZ_H);

  const int NX = 4096 * 1024, NW = 3072 * 1024;
  cvt_f32_to_bf16<<<(NX / 4) / 256, 256, 0, stream>>>((const float4*)x, xb, NX / 4);
  cvt_f32_to_bf16<<<(NW / 4) / 256, 256, 0, stream>>>((const float4*)Wqkv, wb, NW / 4);
  qkv_gemm_wmma<<<dim3(64, 12), 256, 0, stream>>>(xb, wb, bqkv, Qb, Kb, Vt);
  attn_wmma<<<512, 128, 0, stream>>>(Qb, Kb, Vt, out);
}
